// LEAStereo_31464930411131
// MI455X (gfx1250) — matrix operand order repaired and checked by reading, not compiled, not run on hardware
//
#include <hip/hip_runtime.h>
#include <hip/hip_bf16.h>

typedef __attribute__((ext_vector_type(16))) _Float16 v16h;
typedef __attribute__((ext_vector_type(8)))  float    v8f;

// ---- problem constants (from reference: B=2,T=3,C=32,H=88,W=116,D=33) ----
#define BB   2
#define TT   3
#define CC   32
#define HH   88
#define WW   116
#define DD   33
#define HW   (HH*WW)         // 10208
#define CHW  (CC*HW)         // 326656
#define COSTN (DD*HW)        // 336864 cost elements per n
#define OH   260
#define OW   346

// ---- tiling: out tile (d,h,w)=(4,4,14); partial tile with halo = 6x6x16 ----
#define DT 4
#define HT 4
#define WT 14
#define PD 6
#define PH 6
#define PW 16
#define NPTS   (PD*PH*PW)    // 576 partial points
#define MTILES (NPTS/16)     // 36 WMMA row-tiles
#define NDT 9                // ceil(33/4)
#define NHT 22               // 88/4
#define NWT 9                // ceil(116/14)

// Kernel 1: kn2row cost-volume conv.
//   partial[p,tap] = sum_c sig[c,p] * W[c,tap]   (GEMM, WMMA f32_16x16x32_f16)
//   out[q]         = sum_tap partial[q+off(tap), tap]
// sig[c,(d,h,w)] = (c<32 ? x[c,h,w] : y[c-32,h,w-d]) * (w>=d), zero outside
// D/H/W ranges (SAME padding). Only t==2 frames contribute to the output.
__global__ __launch_bounds__(256) void cost_conv_wmma(
    const float* __restrict__ xf, const float* __restrict__ yf,
    const float* __restrict__ wm, float* __restrict__ cost)
{
  __shared__ float part[NPTS][27];   // 62208 B

  int bx = blockIdx.x;
  int wt = bx % NWT; bx /= NWT;
  int ht = bx % NHT; bx /= NHT;
  int dt = bx % NDT; bx /= NDT;
  int n  = bx;                              // 0..1
  int d0 = dt * DT, h0 = ht * HT, w0 = wt * WT;

  int lane = threadIdx.x & 31;
  int wave = threadIdx.x >> 5;
  bool lo16 = (lane < 16);
  int  l16  = lane & 15;

  // only frame t=2 of each batch feeds the output
  const float* xb = xf + (size_t)(3 * n + 2) * CHW;
  const float* yb = yf + (size_t)(3 * n + 2) * CHW;

  // ---- B fragments: B[k=c, n=tap], constant for the whole kernel ----
  // 32x16 f16 B layout: col = lane%16; K = (e<8) ? kbase+e : 8+kbase+e, kbase = (lane<16 ? 0 : 8)
  // Branchless tail handling: taps 27..31 load a clamped (valid) address and
  // then select 0 -> straight-line load clause + v_cndmask, no exec branches.
  v16h bfrag[2][2];
  #pragma unroll
  for (int kc = 0; kc < 2; ++kc) {
    #pragma unroll
    for (int nc = 0; nc < 2; ++nc) {
      int tap = nc * 16 + l16;
      int tapc = (tap < 27) ? tap : 26;     // clamped, always in-bounds
      #pragma unroll
      for (int e = 0; e < 16; ++e) {
        int K = (e < 8) ? (e + (lo16 ? 0 : 8)) : (e + (lo16 ? 8 : 16));
        int c = kc * 32 + K;
        float v = wm[c * 27 + tapc];
        v = (tap < 27) ? v : 0.0f;
        bfrag[kc][nc][e] = (_Float16)v;
      }
    }
  }

  // ---- GEMM phase: each wave handles M-tiles wave, wave+8, ... ----
  for (int tile = wave; tile < MTILES; tile += 8) {
    int pt = tile * 16 + l16;               // partial point = A row (lane%16)
    int pw = pt & 15;                       // PW==16 -> lanes 0..15 coalesce in w
    int ph = (pt >> 4) % PH;
    int pd = pt / (PH * PW);
    int gd = d0 + pd - 1;
    int gh = h0 + ph - 1;
    int gw = w0 + pw - 1;
    bool valid = (gd >= 0) & (gd < DD) & (gh >= 0) & (gh < HH)
               & (gw >= 0) & (gw < WW) & (gw >= gd);
    int xoff = gh * WW + gw;
    int yoff = gh * WW + (gw - gd);

    // prefetch next tile's x row into cache (gfx1250 global_prefetch path)
    if (tile + 8 < MTILES)
      __builtin_prefetch(xb + xoff + 8 * 16, 0, 1);

    // 16x32 f16 A layout: row = lane%16;
    //   K = (e<8) ? kbase+e : 8+kbase+e, kbase = (lane<16 ? 0 : 8)
    v16h a0 = {};  // channels 0..31  (left half: x * mask)
    v16h a1 = {};  // channels 32..63 (right half: y shifted * mask)
    int kbase = lo16 ? 0 : 8;
    if (valid) {
      #pragma unroll
      for (int e = 0; e < 16; ++e) {
        int K = (e < 8) ? (kbase + e) : (8 + kbase + e);
        a0[e] = (_Float16)xb[K * HW + xoff];
        a1[e] = (_Float16)yb[K * HW + yoff];
      }
    }

    v8f c0 = {}, c1 = {};
    c0 = __builtin_amdgcn_wmma_f32_16x16x32_f16(false, a0, false, bfrag[0][0], (short)0, c0, false, false);
    c0 = __builtin_amdgcn_wmma_f32_16x16x32_f16(false, a1, false, bfrag[1][0], (short)0, c0, false, false);
    c1 = __builtin_amdgcn_wmma_f32_16x16x32_f16(false, a0, false, bfrag[0][1], (short)0, c1, false, false);
    c1 = __builtin_amdgcn_wmma_f32_16x16x32_f16(false, a1, false, bfrag[1][1], (short)0, c1, false, false);

    // D layout: row = r + (lane<16 ? 0 : 8), col = lane%16
    int mb = tile * 16 + (lo16 ? 0 : 8);
    #pragma unroll
    for (int r = 0; r < 8; ++r) {
      part[mb + r][l16] = c0[r];
      if (l16 + 16 < 27) part[mb + r][l16 + 16] = c1[r];
    }
  }
  __syncthreads();

  // ---- tap-shift sum: out[q] = sum over 27 taps of shifted partials ----
  int q = threadIdx.x;
  if (q < DT * HT * WT) {
    int ow_ = q % WT;
    int oh_ = (q / WT) % HT;
    int od_ = q / (WT * HT);
    int d = d0 + od_, h = h0 + oh_, w = w0 + ow_;
    if (d < DD && w < WW) {                 // h always < 88 (22*4 exact)
      float acc = 0.0f;
      #pragma unroll
      for (int kd = 0; kd < 3; ++kd)
        #pragma unroll
        for (int kh = 0; kh < 3; ++kh)
          #pragma unroll
          for (int kw = 0; kw < 3; ++kw)
            acc += part[((od_ + kd) * PH + (oh_ + kh)) * PW + (ow_ + kw)]
                       [kd * 9 + kh * 3 + kw];
      cost[(size_t)n * COSTN + d * HW + h * WW + w] = acc;
    }
  }
}

// Kernel 2: align-corners bilinear resize [2,33,88,116] -> [2,33,260,346]
__global__ __launch_bounds__(256) void resize_bilinear_ac(
    const float* __restrict__ cost, float* __restrict__ out, int total)
{
  int idx = blockIdx.x * blockDim.x + threadIdx.x;
  if (idx >= total) return;
  int ox = idx % OW;
  int t  = idx / OW;
  int oy = t % OH; t /= OH;
  int dch = t % DD; t /= DD;
  int n = t;

  const float sy = (float)(HH - 1) / (float)(OH - 1);
  const float sx = (float)(WW - 1) / (float)(OW - 1);
  float ys = oy * sy;
  float xs = ox * sx;
  int y0 = (int)ys; y0 = (y0 > HH - 1) ? (HH - 1) : y0;
  int x0 = (int)xs; x0 = (x0 > WW - 1) ? (WW - 1) : x0;
  int y1 = (y0 + 1 > HH - 1) ? (HH - 1) : (y0 + 1);
  int x1 = (x0 + 1 > WW - 1) ? (WW - 1) : (x0 + 1);
  float wy = ys - (float)y0;
  float wx = xs - (float)x0;

  const float* base = cost + (size_t)n * COSTN + (size_t)dch * HW;
  float r0 = base[y0 * WW + x0] * (1.0f - wx) + base[y0 * WW + x1] * wx;
  float r1 = base[y1 * WW + x0] * (1.0f - wx) + base[y1 * WW + x1] * wx;
  out[idx] = r0 * (1.0f - wy) + r1 * wy;
}

extern "C" void kernel_launch(void* const* d_in, const int* in_sizes, int n_in,
                              void* d_out, int out_size, void* d_ws, size_t ws_size,
                              hipStream_t stream) {
  const float* xf = (const float*)d_in[0];   // [2,3,32,88,116] f32
  const float* yf = (const float*)d_in[1];   // [2,3,32,88,116] f32
  const float* wm = (const float*)d_in[2];   // [1,64,3,3,3]    f32
  float* out  = (float*)d_out;               // [2,33,260,346]  f32
  float* cost = (float*)d_ws;                // scratch: 2*33*88*116 f32 = 2.7 MB

  dim3 g1(BB * NDT * NHT * NWT);             // 3564 workgroups
  cost_conv_wmma<<<g1, 256, 0, stream>>>(xf, yf, wm, cost);

  int total = BB * DD * OH * OW;             // 5,937,360
  resize_bilinear_ac<<<(total + 255) / 256, 256, 0, stream>>>(cost, out, total);
}
